// MultiHeadAttention_56264071578084
// MI455X (gfx1250) — matrix operand order repaired and checked by reading, not compiled, not run on hardware
//
#include <hip/hip_runtime.h>
#include <hip/hip_bf16.h>

// ---------------------------------------------------------------------------
// MHA block for MI455X (gfx1250, wave32).  bf16 WMMA (v_wmma_f32_16x16x32_bf16)
// for all GEMMs, flash-style streaming softmax (never materializes the 512MB
// score tensor), fused FC+residual+LayerNorm epilogue.  Software-pipelined
// fragment loads so WMMAs don't drain LOADcnt to 0.
//
// B=4, L=2048, D_MODEL=512, H=8, DK=DV=32.
// ---------------------------------------------------------------------------

typedef __bf16 bf16;
typedef __attribute__((ext_vector_type(16))) bf16  v16bf;
typedef __attribute__((ext_vector_type(8)))  bf16  v8bf;
typedef __attribute__((ext_vector_type(8)))  float v8f;

#define B_    4
#define L_    2048
#define DM_   512
#define H_    8
#define DK_   32
#define INNER 256          // H_*DK_
#define M_TOT 8192         // B_*L_

// ---------------- fragment loaders (CDNA5 wave32 WMMA, contiguous-k order in A and B) -------------
// A-matrix 16x32 bf16: lane l -> row m0 + (l&15); per-lane data = one
// contiguous 32B chunk of k at offset 16*(l>>4).
__device__ __forceinline__ v16bf load_a_frag(const bf16* __restrict__ base,
                                             int ld, int m0, int k0,
                                             int l16, int lh) {
  const bf16* p = base + (size_t)(m0 + l16) * ld + k0 + 16 * lh;
  v8bf a0 = *(const v8bf*)(p);
  v8bf a1 = *(const v8bf*)(p + 8);
  v16bf A;
#pragma unroll
  for (int i = 0; i < 8; ++i) { A[i] = a0[i]; A[8 + i] = a1[i]; }
  return A;
}

// B-matrix 32x16 bf16: lane l -> column n = nb + (l&15); per-lane data = one
// contiguous 32B chunk of k at offset 16*(l>>4).  Memory must be laid out
// "n-major, k-fast" (i.e. W[n][k] row-major), which all our operands are.
__device__ __forceinline__ v16bf load_b_frag(const bf16* __restrict__ Wn,
                                             int ld, int nb, int k0,
                                             int l16, int lh) {
  return *(const v16bf*)(Wn + (size_t)(nb + l16) * ld + k0 + 16 * lh);
}

__device__ __forceinline__ float redmax16(float t) {
  t = fmaxf(t, __shfl_xor(t, 1, 32));
  t = fmaxf(t, __shfl_xor(t, 2, 32));
  t = fmaxf(t, __shfl_xor(t, 4, 32));
  t = fmaxf(t, __shfl_xor(t, 8, 32));
  return t;   // all 16 lanes of the half-group hold the row max
}
__device__ __forceinline__ float redsum16(float t) {
  t += __shfl_xor(t, 1, 32);
  t += __shfl_xor(t, 2, 32);
  t += __shfl_xor(t, 4, 32);
  t += __shfl_xor(t, 8, 32);
  return t;
}

// ---------------------------- fp32 -> bf16 convert -------------------------
__global__ void cvt_bf16_kernel(const float* __restrict__ src,
                                bf16* __restrict__ dst, int n) {
  int i = blockIdx.x * blockDim.x + threadIdx.x;
  if (i < n) dst[i] = (bf16)src[i];
}

// ------------------------- QKV projection + head scatter -------------------
// P[m, c] = sum_k X[m,k] * W[c,k] + bias[c], then scatter per reference
// reshape: c = d*8 + h  ->  head h, dim d.
// mode 0: Qh[b,h,l,d]  (pre-scaled by 1/sqrt(DK))
// mode 1: Kh[b,h,l,d]
// mode 2: Vt[b,h,d,l]  (transposed so PV B-frags load contiguously)
__global__ void proj_kernel(const bf16* __restrict__ X,
                            const bf16* __restrict__ W,
                            const float* __restrict__ bias,
                            bf16* __restrict__ dst,
                            int mode, float scale) {
  const int lane = threadIdx.x & 31, l16 = lane & 15, lh = lane >> 4;
  const int wv = threadIdx.x >> 5;            // 4 waves, 64 cols each
  const int m0 = blockIdx.x * 16;

  v8f c[4] = {};
  // software pipeline: current fragments in A/Bf, next k-step prefetched
  v16bf A = load_a_frag(X, DM_, m0, 0, l16, lh);
  v16bf Bf[4];
#pragma unroll
  for (int f = 0; f < 4; ++f)
    Bf[f] = load_b_frag(W, DM_, wv * 64 + f * 16, 0, l16, lh);

#pragma unroll 2
  for (int k0 = 0; k0 < DM_; k0 += 32) {
    const int kn = (k0 + 32 < DM_) ? (k0 + 32) : k0;   // clamp (redundant last)
    v16bf An = load_a_frag(X, DM_, m0, kn, l16, lh);
    v16bf Bn[4];
#pragma unroll
    for (int f = 0; f < 4; ++f)
      Bn[f] = load_b_frag(W, DM_, wv * 64 + f * 16, kn, l16, lh);
#pragma unroll
    for (int f = 0; f < 4; ++f)
      c[f] = __builtin_amdgcn_wmma_f32_16x16x32_bf16(false, A, false, Bf[f],
                                                     (short)0, c[f],
                                                     false, false);
    A = An;
#pragma unroll
    for (int f = 0; f < 4; ++f) Bf[f] = Bn[f];
  }

#pragma unroll
  for (int f = 0; f < 4; ++f) {
#pragma unroll
    for (int r = 0; r < 8; ++r) {
      int row = m0 + r + 8 * lh;              // global m = b*L + l
      int col = wv * 64 + f * 16 + l16;       // 0..255
      float v = (c[f][r] + bias[col]) * scale;
      int b = row >> 11, l = row & (L_ - 1);
      int h = col & 7, d = col >> 3;
      size_t idx;
      if (mode == 2) idx = ((size_t)(b * H_ + h) * DK_ + d) * L_ + l;
      else           idx = ((size_t)(b * H_ + h) * L_ + l) * DK_ + d;
      dst[idx] = (bf16)v;
    }
  }
}

// -------------------- flash attention: one wave = 16 q rows ----------------
__global__ void attn_kernel(const bf16* __restrict__ Qh,
                            const bf16* __restrict__ Kh,
                            const bf16* __restrict__ Vt,
                            bf16* __restrict__ Attn) {
  const int lane = threadIdx.x & 31, l16 = lane & 15, lh = lane >> 4;
  const int bh = blockIdx.x >> 7;             // 0..31  (b*8 + h)
  const int qt = blockIdx.x & 127;
  const int q0 = qt * 16;

  const bf16* Q = Qh + (size_t)bh * L_ * DK_;
  const bf16* K = Kh + (size_t)bh * L_ * DK_;
  const bf16* V = Vt + (size_t)bh * DK_ * L_;

  __shared__ bf16 Pt[16 * 32];                // P tile, row-major [m][kk]

  v16bf QA = load_a_frag(Q, DK_, q0, 0, l16, lh);   // Q already scaled
  v8f O0 = {}, O1 = {};
  float mrow[8], lrow[8];
#pragma unroll
  for (int r = 0; r < 8; ++r) { mrow[r] = -3.0e30f; lrow[r] = 0.f; }
  const v8f zero = {};

  // preload first K fragments
  v16bf KB0 = *(const v16bf*)(K + (size_t)l16 * DK_ + 16 * lh);
  v16bf KB1 = *(const v16bf*)(K + (size_t)(16 + l16) * DK_ + 16 * lh);

  for (int kb = 0; kb < L_; kb += 32) {
    // scores: S = Q(16x32) * K^T  -> two 16-key fragments
    v8f S0 = __builtin_amdgcn_wmma_f32_16x16x32_bf16(false, QA, false, KB0,
                                                     (short)0, zero, false, false);
    v8f S1 = __builtin_amdgcn_wmma_f32_16x16x32_bf16(false, QA, false, KB1,
                                                     (short)0, zero, false, false);
    // issue V loads (this iter) + K loads (next iter) early so their latency
    // hides behind the softmax VALU work and the LDS transpose below
    v16bf VB0 = *(const v16bf*)(V + (size_t)l16 * L_ + kb + 16 * lh);
    v16bf VB1 = *(const v16bf*)(V + (size_t)(16 + l16) * L_ + kb + 16 * lh);
    const int kn = (kb + 32 < L_) ? (kb + 32) : kb;
    v16bf KN0 = *(const v16bf*)(K + (size_t)(kn + l16) * DK_ + 16 * lh);
    v16bf KN1 = *(const v16bf*)(K + (size_t)(kn + 16 + l16) * DK_ + 16 * lh);
    // prefetch two tiles ahead (global_prefetch_b8); streamed K/V data
    const int kp = (kb + 64 < L_) ? (kb + 64) : kb;
    __builtin_prefetch(K + (size_t)(kp + l16) * DK_, 0, 1);
    __builtin_prefetch(V + (size_t)l16 * L_ + kp, 0, 1);

    // online softmax, rows r+8*lh live across the 16-lane half-group
#pragma unroll
    for (int r = 0; r < 8; ++r) {
      float tmax = redmax16(fmaxf(S0[r], S1[r]));
      float mn   = fmaxf(mrow[r], tmax);
      float al   = __expf(mrow[r] - mn);
      mrow[r] = mn;
      float p0 = __expf(S0[r] - mn);
      float p1 = __expf(S1[r] - mn);
      S0[r] = p0; S1[r] = p1;
      lrow[r] = lrow[r] * al + redsum16(p0 + p1);
      O0[r] *= al; O1[r] *= al;
    }
    // C-layout -> A-layout transpose of P through LDS (1KB, single wave)
#pragma unroll
    for (int r = 0; r < 8; ++r) {
      Pt[(r + 8 * lh) * 32 + l16]      = (bf16)S0[r];
      Pt[(r + 8 * lh) * 32 + 16 + l16] = (bf16)S1[r];
    }
    v8bf a0 = *(const v8bf*)(Pt + l16 * 32 + 16 * lh);
    v8bf a1 = *(const v8bf*)(Pt + l16 * 32 + 8 + 16 * lh);
    v16bf PA;
#pragma unroll
    for (int i = 0; i < 8; ++i) { PA[i] = a0[i]; PA[8 + i] = a1[i]; }
    // O += P(16x32) * V(32x32): Vt is [d][l], so B-frags are contiguous
    O0 = __builtin_amdgcn_wmma_f32_16x16x32_bf16(false, PA, false, VB0,
                                                 (short)0, O0, false, false);
    O1 = __builtin_amdgcn_wmma_f32_16x16x32_bf16(false, PA, false, VB1,
                                                 (short)0, O1, false, false);
    KB0 = KN0; KB1 = KN1;
  }
  // epilogue: normalize and scatter to Attn[b, l, h*32 + d]
  const int b = bh >> 3, h = bh & 7;
#pragma unroll
  for (int r = 0; r < 8; ++r) {
    float inv = 1.0f / lrow[r];
    int row = q0 + r + 8 * lh;
    size_t base = ((size_t)b * L_ + row) * INNER + h * 32;
    Attn[base + l16]      = (bf16)(O0[r] * inv);
    Attn[base + 16 + l16] = (bf16)(O1[r] * inv);
  }
}

// ------------------- FC + bias + residual + LayerNorm ----------------------
__global__ void fc_ln_kernel(const bf16* __restrict__ Attn,
                             const bf16* __restrict__ Wfc,
                             const float* __restrict__ bfc,
                             const float* __restrict__ qres,
                             const float* __restrict__ gamma,
                             const float* __restrict__ beta,
                             float* __restrict__ out) {
  const int lane = threadIdx.x & 31, l16 = lane & 15, lh = lane >> 4;
  const int wv = threadIdx.x >> 5;            // 8 waves, 64 cols each
  const int m0 = blockIdx.x * 16;

  __shared__ float xb[16 * DM_];              // 32 KB
  __shared__ float rs[16][17], rs2[16][17];
  __shared__ float mu_s[16], rstd_s[16];

  v8f c[4] = {};
  v16bf A = load_a_frag(Attn, INNER, m0, 0, l16, lh);
  v16bf Bf[4];
#pragma unroll
  for (int f = 0; f < 4; ++f)
    Bf[f] = load_b_frag(Wfc, INNER, wv * 64 + f * 16, 0, l16, lh);

#pragma unroll 2
  for (int k0 = 0; k0 < INNER; k0 += 32) {
    const int kn = (k0 + 32 < INNER) ? (k0 + 32) : k0;
    v16bf An = load_a_frag(Attn, INNER, m0, kn, l16, lh);
    v16bf Bn[4];
#pragma unroll
    for (int f = 0; f < 4; ++f)
      Bn[f] = load_b_frag(Wfc, INNER, wv * 64 + f * 16, kn, l16, lh);
#pragma unroll
    for (int f = 0; f < 4; ++f)
      c[f] = __builtin_amdgcn_wmma_f32_16x16x32_bf16(false, A, false, Bf[f],
                                                     (short)0, c[f],
                                                     false, false);
    A = An;
#pragma unroll
    for (int f = 0; f < 4; ++f) Bf[f] = Bn[f];
  }

#pragma unroll
  for (int f = 0; f < 4; ++f) {
#pragma unroll
    for (int r = 0; r < 8; ++r) {
      int row = r + 8 * lh;
      int col = wv * 64 + f * 16 + l16;
      float v = c[f][r] + bfc[col] + qres[(size_t)(m0 + row) * DM_ + col];
      xb[row * DM_ + col] = v;
    }
  }
  __syncthreads();

  // LayerNorm over 512 cols: 16 threads per row, 32 elems each
  const int row = threadIdx.x & 15, seg = threadIdx.x >> 4;
  float s = 0.f, s2 = 0.f;
#pragma unroll 8
  for (int j = 0; j < 32; ++j) {
    float x = xb[row * DM_ + seg * 32 + j];
    s += x; s2 += x * x;
  }
  rs[row][seg] = s; rs2[row][seg] = s2;
  __syncthreads();
  if (seg == 0) {
    float a = 0.f, b2 = 0.f;
#pragma unroll
    for (int j = 0; j < 16; ++j) { a += rs[row][j]; b2 += rs2[row][j]; }
    float mu = a * (1.0f / DM_);
    float var = b2 * (1.0f / DM_) - mu * mu;
    mu_s[row] = mu;
    rstd_s[row] = rsqrtf(var + 1e-5f);
  }
  __syncthreads();
  float mu = mu_s[row], rv = rstd_s[row];
#pragma unroll 8
  for (int j = 0; j < 32; ++j) {
    int col = seg * 32 + j;
    float x = xb[row * DM_ + col];
    out[(size_t)(m0 + row) * DM_ + col] = gamma[col] * (x - mu) * rv + beta[col];
  }
}

// ---------------------------------------------------------------------------
extern "C" void kernel_launch(void* const* d_in, const int* in_sizes, int n_in,
                              void* d_out, int out_size, void* d_ws, size_t ws_size,
                              hipStream_t stream) {
  const float* q   = (const float*)d_in[0];
  const float* k   = (const float*)d_in[1];
  const float* v   = (const float*)d_in[2];
  const float* Wq  = (const float*)d_in[3];
  const float* bq  = (const float*)d_in[4];
  const float* Wk  = (const float*)d_in[5];
  const float* bk  = (const float*)d_in[6];
  const float* Wv  = (const float*)d_in[7];
  const float* bv  = (const float*)d_in[8];
  const float* Wfc = (const float*)d_in[9];
  const float* bfc = (const float*)d_in[10];
  const float* gam = (const float*)d_in[11];
  const float* bet = (const float*)d_in[12];
  float* out = (float*)d_out;

  // workspace partition (bytes); total ~41 MB
  char* ws = (char*)d_ws;
  const size_t SZ_X  = (size_t)M_TOT * DM_ * 2;        // 8 MB
  const size_t SZ_W  = (size_t)INNER * DM_ * 2;        // 256 KB
  const size_t SZ_H  = (size_t)B_ * H_ * L_ * DK_ * 2; // 4 MB
  bf16* Xq  = (bf16*)(ws);               ws += SZ_X;
  bf16* Xk  = (bf16*)(ws);               ws += SZ_X;
  bf16* Xv  = (bf16*)(ws);               ws += SZ_X;
  bf16* Wqb = (bf16*)(ws);               ws += SZ_W;
  bf16* Wkb = (bf16*)(ws);               ws += SZ_W;
  bf16* Wvb = (bf16*)(ws);               ws += SZ_W;
  bf16* Wfb = (bf16*)(ws);               ws += SZ_W;
  bf16* Qh  = (bf16*)(ws);               ws += SZ_H;
  bf16* Kh  = (bf16*)(ws);               ws += SZ_H;
  bf16* Vt  = (bf16*)(ws);               ws += SZ_H;
  bf16* At  = (bf16*)(ws);               ws += (size_t)M_TOT * INNER * 2;

  const int NX = M_TOT * DM_;      // 4,194,304
  const int NW = INNER * DM_;      // 131,072
  cvt_bf16_kernel<<<NX / 256, 256, 0, stream>>>(q,   Xq,  NX);
  cvt_bf16_kernel<<<NX / 256, 256, 0, stream>>>(k,   Xk,  NX);
  cvt_bf16_kernel<<<NX / 256, 256, 0, stream>>>(v,   Xv,  NX);
  cvt_bf16_kernel<<<NW / 256, 256, 0, stream>>>(Wq,  Wqb, NW);
  cvt_bf16_kernel<<<NW / 256, 256, 0, stream>>>(Wk,  Wkb, NW);
  cvt_bf16_kernel<<<NW / 256, 256, 0, stream>>>(Wv,  Wvb, NW);
  cvt_bf16_kernel<<<NW / 256, 256, 0, stream>>>(Wfc, Wfb, NW);

  const float qscale = 0.17677669529663687f;  // 1/sqrt(DK), folded into Q proj
  proj_kernel<<<M_TOT / 16, 128, 0, stream>>>(Xq, Wqb, bq, Qh, 0, qscale);
  proj_kernel<<<M_TOT / 16, 128, 0, stream>>>(Xk, Wkb, bk, Kh, 1, 1.0f);
  proj_kernel<<<M_TOT / 16, 128, 0, stream>>>(Xv, Wvb, bv, Vt, 2, 1.0f);

  attn_kernel<<<B_ * H_ * (L_ / 16), 32, 0, stream>>>(Qh, Kh, Vt, At);

  fc_ln_kernel<<<M_TOT / 16, 256, 0, stream>>>(At, Wfb, bfc, q, gam, bet, out);
}
